// ParallelLorentzEncoder_35923106463885
// MI455X (gfx1250) — compile-verified, matrix operand order checked by reading, not run on hardware
//
#include <hip/hip_runtime.h>
#include <hip/hip_bf16.h>
#include <math.h>

// Problem constants (from reference)
#define NEN 3
#define BB 8
#define TSEQ 1024
#define HDIM 128
#define EDIM 64
#define LLAY 3
#define DDI 256
#define NST 16
#define KCV 4
#define RDT 8
#define MTOT (BB*TSEQ)   // 8192 rows per encoder
#define XPN 64           // x-proj padded width (real N = 40)

typedef __attribute__((ext_vector_type(16))) __bf16 bf16x16;
typedef __attribute__((ext_vector_type(8)))  __bf16 bf16x8;
typedef __attribute__((ext_vector_type(8)))  float  floatx8;

__device__ __forceinline__ __bf16 f2bf(float f) {
  unsigned u = __builtin_bit_cast(unsigned, f);
  unsigned r = (u + 0x7FFFu + ((u >> 16) & 1u)) >> 16;   // round-to-nearest-even
  unsigned short s = (unsigned short)r;
  return __builtin_bit_cast(__bf16, s);
}

// ---------------------------------------------------------------------------
// Batched GEMM: C[e][M][ldc] = A[e][M][K] * W[e][N][K]^T
// A, W already bf16; inner loop = pure b128 loads + v_wmma (no conversion VALU).
// block = 256 threads = 8 waves; wave computes 16 rows x 64 cols.
// grid = (M/128, ldc/64, NEN). OUTBF: store C as bf16 (else fp32).
// ---------------------------------------------------------------------------
template <bool OUTBF>
__global__ void gemm_bf16_wmma(const __bf16* __restrict__ A, long sA,
                               const __bf16* __restrict__ W, long sW,
                               void* __restrict__ C, long sC,
                               int M, int K, int ldc) {
  const int e     = blockIdx.z;
  const __bf16* Ae = A + (long)e * sA;
  const __bf16* We = W + (long)e * sW;

  const int wave = threadIdx.x >> 5;
  const int lane = threadIdx.x & 31;
  const int m0   = blockIdx.x * 128 + wave * 16;
  const int n0   = blockIdx.y * 64;
  if (m0 >= M) return;

  const int ml = lane & 15;          // row (A frag) / col (B frag) within tile
  const int kb = (lane >> 4) * 8;    // K sub-offset per lane half (ISA 16-bit layout)

  floatx8 acc[4];
  #pragma unroll
  for (int t = 0; t < 4; ++t) acc[t] = (floatx8){};

  const __bf16* arow = Ae + (long)(m0 + ml) * K + kb;
  const __bf16* wrow0 = We + (long)(n0 + ml) * K + kb;

  #pragma unroll 2
  for (int kc = 0; kc < K; kc += 32) {
    // A fragment: VGPRs 0..3 = K pairs kb..kb+7, VGPRs 4..7 = 16+kb..16+kb+7
    bf16x8 alo = *(const bf16x8*)(arow + kc);
    bf16x8 ahi = *(const bf16x8*)(arow + kc + 16);
    bf16x16 a = __builtin_shufflevector(alo, ahi, 0,1,2,3,4,5,6,7,8,9,10,11,12,13,14,15);
    #pragma unroll
    for (int t = 0; t < 4; ++t) {
      const __bf16* wrow = wrow0 + (long)(t * 16) * K + kc;
      bf16x8 blo = *(const bf16x8*)(wrow);
      bf16x8 bhi = *(const bf16x8*)(wrow + 16);
      bf16x16 b = __builtin_shufflevector(blo, bhi, 0,1,2,3,4,5,6,7,8,9,10,11,12,13,14,15);
      acc[t] = __builtin_amdgcn_wmma_f32_16x16x32_bf16(
          false, a, false, b, (short)0, acc[t], false, false);
    }
  }

  // C layout (ISA): lanes 0-15 -> M = i, N = lane; lanes 16-31 -> M = 8+i
  #pragma unroll
  for (int t = 0; t < 4; ++t) {
    const int n = n0 + t * 16 + ml;
    #pragma unroll
    for (int i = 0; i < 8; ++i) {
      const int row = m0 + ((lane < 16) ? i : (8 + i));
      if (OUTBF) {
        ((__bf16*)C + (long)e * sC)[(long)row * ldc + n] = f2bf(acc[t][i]);
      } else {
        ((float*)C + (long)e * sC)[(long)row * ldc + n] = acc[t][i];
      }
    }
  }
}

// ---------------------------------------------------------------------------
// Weight conversion fp32 -> bf16 (run each launch; ~1M elements total)
// ---------------------------------------------------------------------------
__global__ void cvt_bf16_kernel(const float* __restrict__ src,
                                __bf16* __restrict__ dst, long n) {
  long i = (long)blockIdx.x * blockDim.x + threadIdx.x;
  if (i < n) dst[i] = f2bf(src[i]);
}

// x-proj weights (NE*L, 40, 256) -> zero-padded (NE*L, 64, 256) bf16
__global__ void cvt_xproj_kernel(const float* __restrict__ src,
                                 __bf16* __restrict__ dst) {
  long i = (long)blockIdx.x * blockDim.x + threadIdx.x;
  const long tot = (long)NEN * LLAY * XPN * DDI;
  if (i >= tot) return;
  int k = (int)(i % DDI);
  int n = (int)((i / DDI) % XPN);
  long g = i / ((long)DDI * XPN);
  dst[i] = (n < 40) ? f2bf(src[(g * 40 + n) * DDI + k]) : f2bf(0.0f);
}

// ---------------------------------------------------------------------------
// Embed: hbf[e][m][hh] = bf16(x[e][m] * in_w[e][hh] + in_b[e][hh])
// ---------------------------------------------------------------------------
__global__ void embed_kernel(const float* __restrict__ x,
                             const float* __restrict__ inw,
                             const float* __restrict__ inb,
                             __bf16* __restrict__ h) {
  long i = (long)blockIdx.x * blockDim.x + threadIdx.x;
  if (i >= (long)NEN * MTOT * HDIM) return;
  int  hh = (int)(i % HDIM);
  long m  = (i / HDIM) % MTOT;
  int  e  = (int)(i / ((long)HDIM * MTOT));
  h[i] = f2bf(x[(long)e * MTOT + m] * inw[e * HDIM + hh] + inb[e * HDIM + hh]);
}

// ---------------------------------------------------------------------------
// Depthwise causal conv (K=4) + bias + SiLU; writes fp32 (for scan) and bf16
// (GEMM A operand for the x-projection).
// ---------------------------------------------------------------------------
__global__ void conv_silu_kernel(const float* __restrict__ xz,
                                 const float* __restrict__ cw, long scw,
                                 const float* __restrict__ cb, long scb,
                                 float* __restrict__ xc,
                                 __bf16* __restrict__ xcbf) {
  long i = (long)blockIdx.x * blockDim.x + threadIdx.x;
  if (i >= (long)NEN * MTOT * DDI) return;
  int  d = (int)(i % DDI);
  long m = (i / DDI) % MTOT;
  int  e = (int)(i / ((long)DDI * MTOT));
  int  t = (int)(m % TSEQ);
  long bbase = m - t;
  const float* xze = xz + (long)e * MTOT * 512;
  const float* wk  = cw + (long)e * scw + (long)d * KCV;
  float s = cb[(long)e * scb + d];
  #pragma unroll
  for (int k = 0; k < KCV; ++k) {
    int tt = t - (KCV - 1) + k;
    if (tt >= 0) s += xze[(bbase + tt) * 512 + d] * wk[k];
  }
  float v = s / (1.0f + __expf(-s));   // SiLU
  xc[i]   = v;
  xcbf[i] = f2bf(v);
}

// ---------------------------------------------------------------------------
// Selective scan. One wave per (e, b, d-pair); lane = (d-sub, n).
// dt projection (K=8) + softplus fused; n-reduction via 16-lane shuffles;
// (y + D*xc)*silu(z) epilogue writes bf16 y (GEMM A operand for out-proj).
// Prefetches (global_prefetch_b8) hide latency of the serial t-loop.
// ---------------------------------------------------------------------------
__global__ void scan_kernel(const float* __restrict__ xdbl,   // [e][m][64]
                            const float* __restrict__ xc,     // [e][m][DDI]
                            const float* __restrict__ xz,     // [e][m][512]
                            const float* __restrict__ dtw, long sdtw,
                            const float* __restrict__ dtb, long sdtb,
                            const float* __restrict__ alog, long salog,
                            const float* __restrict__ Dv, long sDv,
                            __bf16* __restrict__ y) {         // [e][m][DDI]
  const int wave = blockIdx.x * (blockDim.x >> 5) + (threadIdx.x >> 5);
  const int lane = threadIdx.x & 31;
  const int totalWaves = NEN * BB * (DDI / 2);
  if (wave >= totalWaves) return;
  const int e   = wave / (BB * (DDI / 2));
  const int rem = wave % (BB * (DDI / 2));
  const int b   = rem / (DDI / 2);
  const int dp  = rem % (DDI / 2);
  const int d   = dp * 2 + (lane >> 4);
  const int n   = lane & 15;

  const float Acoef = -__expf(alog[(long)e * salog + (long)d * NST + n]);
  const float bias  = dtb[(long)e * sdtb + d];
  const float Dd    = Dv [(long)e * sDv  + d];
  float wreg[RDT];
  #pragma unroll
  for (int r = 0; r < RDT; ++r) wreg[r] = dtw[(long)e * sdtw + (long)d * RDT + r];

  const long base = (long)e * MTOT + (long)b * TSEQ;
  float hstate = 0.0f;

  for (int t = 0; t < TSEQ; ++t) {
    const float* xrow = xdbl + (base + t) * XPN;
    if (t + 8 < TSEQ) {
      __builtin_prefetch(xdbl + (base + t + 8) * XPN, 0, 0);
      __builtin_prefetch(xc + (base + t + 8) * DDI + d, 0, 0);
    }
    float dtr = bias;
    #pragma unroll
    for (int r = 0; r < RDT; ++r) dtr += xrow[r] * wreg[r];
    const float dt  = (dtr > 20.0f) ? dtr : log1pf(__expf(dtr));  // softplus
    const float Bm  = xrow[RDT + n];
    const float Cm  = xrow[RDT + NST + n];
    const float xcv = xc[(base + t) * DDI + d];
    hstate = __expf(dt * Acoef) * hstate + dt * Bm * xcv;
    float part = hstate * Cm;
    part += __shfl_xor(part, 8, 16);
    part += __shfl_xor(part, 4, 16);
    part += __shfl_xor(part, 2, 16);
    part += __shfl_xor(part, 1, 16);
    if (n == 0) {
      const float z  = xz[(base + t) * 512 + DDI + d];
      const float yo = (part + Dd * xcv) * (z / (1.0f + __expf(-z)));
      y[(base + t) * DDI + d] = f2bf(yo);
    }
  }
}

// ---------------------------------------------------------------------------
// Final: tanh + bias -> tangents; expmap0+projx -> h; logmap0 -> sum -> combined.
// One wave per (b,t); lane owns dims {lane, lane+32} of the 64-dim vectors.
// ---------------------------------------------------------------------------
__global__ void final_kernel(const float* __restrict__ tang,   // [e][m][64]
                             const float* __restrict__ outb,   // [e][64]
                             float* __restrict__ out) {
  const int wave = blockIdx.x * (blockDim.x >> 5) + (threadIdx.x >> 5);
  const int lane = threadIdx.x & 31;
  if (wave >= MTOT) return;
  const long m = wave;
  const float EPSf = 1e-7f;

  float* outTan = out;                                        // 3 * MTOT*64
  float* outH   = out + (long)3 * MTOT * EDIM;                // 3 * MTOT*65
  float* outCT  = outH + (long)3 * MTOT * (EDIM + 1);         // MTOT*64
  float* outCH  = outCT + (long)MTOT * EDIM;                  // MTOT*65

  float uc0 = 0.0f, uc1 = 0.0f;

  for (int e = 0; e < NEN; ++e) {
    const long rbase = ((long)e * MTOT + m) * EDIM;
    float t0 = tanhf(tang[rbase + lane]      + outb[e * EDIM + lane]);
    float t1 = tanhf(tang[rbase + lane + 32] + outb[e * EDIM + lane + 32]);
    outTan[rbase + lane]      = t0;
    outTan[rbase + lane + 32] = t1;

    // expmap0 (SK=1) + projx
    float n2 = t0 * t0 + t1 * t1;
    for (int off = 16; off; off >>= 1) n2 += __shfl_xor(n2, off, 32);
    const float ns = fmaxf(sqrtf(n2), EPSf);
    const float s  = sinhf(ns) / ns;
    const float xs0 = s * t0, xs1 = s * t1;
    float q2 = xs0 * xs0 + xs1 * xs1;
    for (int off = 16; off; off >>= 1) q2 += __shfl_xor(q2, off, 32);
    const float x0 = sqrtf(1.0f + q2);

    float* hrow = outH + ((long)e * MTOT + m) * (EDIM + 1);
    if (lane == 0) hrow[0] = x0;
    hrow[1 + lane]      = xs0;
    hrow[1 + lane + 32] = xs1;

    // logmap0
    const float xq  = fmaxf(x0, 1.0f + EPSf);
    const float dd  = logf(xq + sqrtf(xq * xq - 1.0f));   // arccosh
    const float nxs = fmaxf(sqrtf(q2), EPSf);
    uc0 += dd * xs0 / nxs;
    uc1 += dd * xs1 / nxs;
  }

  float* ctrow = outCT + m * EDIM;
  ctrow[lane]      = uc0;
  ctrow[lane + 32] = uc1;

  float n2 = uc0 * uc0 + uc1 * uc1;
  for (int off = 16; off; off >>= 1) n2 += __shfl_xor(n2, off, 32);
  const float ns = fmaxf(sqrtf(n2), EPSf);
  const float s  = sinhf(ns) / ns;
  const float xs0 = s * uc0, xs1 = s * uc1;
  float q2 = xs0 * xs0 + xs1 * xs1;
  for (int off = 16; off; off >>= 1) q2 += __shfl_xor(q2, off, 32);

  float* chrow = outCH + m * (EDIM + 1);
  if (lane == 0) chrow[0] = sqrtf(1.0f + q2);
  chrow[1 + lane]      = xs0;
  chrow[1 + lane + 32] = xs1;
}

// ---------------------------------------------------------------------------
extern "C" void kernel_launch(void* const* d_in, const int* in_sizes, int n_in,
                              void* d_out, int out_size, void* d_ws, size_t ws_size,
                              hipStream_t stream) {
  (void)in_sizes; (void)n_in; (void)out_size; (void)ws_size;
  const float* x      = (const float*)d_in[0];
  const float* einw   = (const float*)d_in[1];
  const float* einb   = (const float*)d_in[2];
  const float* m_inw  = (const float*)d_in[3];   // (NE,L,512,128)
  const float* m_cw   = (const float*)d_in[4];   // (NE,L,256,4)
  const float* m_cb   = (const float*)d_in[5];   // (NE,L,256)
  const float* m_xp   = (const float*)d_in[6];   // (NE,L,40,256)
  const float* m_dtw  = (const float*)d_in[7];   // (NE,L,256,8)
  const float* m_dtb  = (const float*)d_in[8];   // (NE,L,256)
  const float* m_alog = (const float*)d_in[9];   // (NE,L,256,16)
  const float* m_D    = (const float*)d_in[10];  // (NE,L,256)
  const float* m_ow   = (const float*)d_in[11];  // (NE,L,128,256)
  const float* eow    = (const float*)d_in[12];  // (NE,64,128)
  const float* eob    = (const float*)d_in[13];  // (NE,64)
  float* out = (float*)d_out;

  // ---- workspace layout (byte offsets, 256B aligned) ----
  char* p = (char*)d_ws;
  auto alloc = [&](size_t bytes) -> char* {
    char* r = p; p += (bytes + 255) & ~(size_t)255; return r;
  };
  float*  XZ    = (float*) alloc((size_t)NEN * MTOT * 512 * 4);
  float*  XC    = (float*) alloc((size_t)NEN * MTOT * DDI * 4);
  float*  XDBL  = (float*) alloc((size_t)NEN * MTOT * XPN * 4);
  float*  TANG  = (float*) alloc((size_t)NEN * MTOT * EDIM * 4);
  __bf16* Hbf0  = (__bf16*)alloc((size_t)NEN * MTOT * HDIM * 2);
  __bf16* Hbf1  = (__bf16*)alloc((size_t)NEN * MTOT * HDIM * 2);
  __bf16* XCbf  = (__bf16*)alloc((size_t)NEN * MTOT * DDI * 2);
  __bf16* Ybf   = (__bf16*)alloc((size_t)NEN * MTOT * DDI * 2);
  __bf16* Winb  = (__bf16*)alloc((size_t)NEN * LLAY * 2 * DDI * HDIM * 2);
  __bf16* Wxpb  = (__bf16*)alloc((size_t)NEN * LLAY * XPN * DDI * 2);
  __bf16* Wowb  = (__bf16*)alloc((size_t)NEN * LLAY * HDIM * DDI * 2);
  __bf16* Weob  = (__bf16*)alloc((size_t)NEN * EDIM * HDIM * 2);

  // ---- weight conversion (bf16, padded x-proj) ----
  {
    long n1 = (long)NEN * LLAY * 2 * DDI * HDIM;
    cvt_bf16_kernel<<<dim3((unsigned)((n1 + 255) / 256)), dim3(256), 0, stream>>>(m_inw, Winb, n1);
    long n2 = (long)NEN * LLAY * XPN * DDI;
    cvt_xproj_kernel<<<dim3((unsigned)((n2 + 255) / 256)), dim3(256), 0, stream>>>(m_xp, Wxpb);
    long n3 = (long)NEN * LLAY * HDIM * DDI;
    cvt_bf16_kernel<<<dim3((unsigned)((n3 + 255) / 256)), dim3(256), 0, stream>>>(m_ow, Wowb, n3);
    long n4 = (long)NEN * EDIM * HDIM;
    cvt_bf16_kernel<<<dim3((unsigned)((n4 + 255) / 256)), dim3(256), 0, stream>>>(eow, Weob, n4);
  }

  // ---- embed ----
  {
    long tot = (long)NEN * MTOT * HDIM;
    embed_kernel<<<dim3((unsigned)((tot + 255) / 256)), dim3(256), 0, stream>>>(
        x, einw, einb, Hbf0);
  }

  __bf16* hcur = Hbf0;
  __bf16* hnext = Hbf1;
  for (int l = 0; l < LLAY; ++l) {
    // in-proj: (8192 x 128)bf16 x (512 x 128)bf16^T -> xz fp32 (8192 x 512)
    gemm_bf16_wmma<false><<<dim3(64, 8, NEN), dim3(256), 0, stream>>>(
        hcur, (long)MTOT * HDIM,
        Winb + (long)l * 2 * DDI * HDIM, (long)LLAY * 2 * DDI * HDIM,
        XZ, (long)MTOT * 512,
        MTOT, HDIM, 512);
    // depthwise conv + SiLU -> xc fp32 + bf16
    {
      long tot = (long)NEN * MTOT * DDI;
      conv_silu_kernel<<<dim3((unsigned)((tot + 255) / 256)), dim3(256), 0, stream>>>(
          XZ, m_cw + (long)l * DDI * KCV, (long)LLAY * DDI * KCV,
          m_cb + (long)l * DDI, (long)LLAY * DDI, XC, XCbf);
    }
    // x-proj: (8192 x 256)bf16 x (64 x 256)bf16^T -> xdbl fp32 (8192 x 64)
    gemm_bf16_wmma<false><<<dim3(64, 1, NEN), dim3(256), 0, stream>>>(
        XCbf, (long)MTOT * DDI,
        Wxpb + (long)l * XPN * DDI, (long)LLAY * XPN * DDI,
        XDBL, (long)MTOT * XPN,
        MTOT, DDI, XPN);
    // selective scan (+ dt proj + epilogue) -> y bf16
    scan_kernel<<<dim3(384), dim3(256), 0, stream>>>(
        XDBL, XC, XZ,
        m_dtw + (long)l * DDI * RDT, (long)LLAY * DDI * RDT,
        m_dtb + (long)l * DDI, (long)LLAY * DDI,
        m_alog + (long)l * DDI * NST, (long)LLAY * DDI * NST,
        m_D + (long)l * DDI, (long)LLAY * DDI,
        Ybf);
    // out-proj: (8192 x 256)bf16 x (128 x 256)bf16^T -> h bf16 (8192 x 128)
    gemm_bf16_wmma<true><<<dim3(64, 2, NEN), dim3(256), 0, stream>>>(
        Ybf, (long)MTOT * DDI,
        Wowb + (long)l * HDIM * DDI, (long)LLAY * HDIM * DDI,
        hnext, (long)MTOT * HDIM,
        MTOT, DDI, HDIM);
    __bf16* tmp = hcur; hcur = hnext; hnext = tmp;
  }

  // Encoder output projection: (8192 x 128)bf16 x (64 x 128)bf16^T -> tang fp32
  gemm_bf16_wmma<false><<<dim3(64, 1, NEN), dim3(256), 0, stream>>>(
      hcur, (long)MTOT * HDIM,
      Weob, (long)EDIM * HDIM,
      TANG, (long)MTOT * EDIM,
      MTOT, HDIM, EDIM);

  // tanh + Lorentz maps + combine -> all 8 outputs
  final_kernel<<<dim3(1024), dim3(256), 0, stream>>>(TANG, eob, out);
}
